// BasicBlock_17806934409817
// MI455X (gfx1250) — matrix operand order repaired and checked by reading, not compiled, not run on hardware
//
#include <hip/hip_runtime.h>
#include <hip/hip_bf16.h>
#include <math.h>
#include <cstdint>

// ---------------------------------------------------------------------------
// MI455X (gfx1250) WeightNet BasicBlock.
// Heavy work = 2x implicit-GEMM 3x3 convs on v_wmma_f32_16x16x32_f16,
// f16 in / f32 accum, double-buffered LDS tiles filled with CDNA5
// global_load_async_to_lds_b128 (ASYNCcnt), wave32, 8 waves/block.
// ---------------------------------------------------------------------------

typedef __attribute__((ext_vector_type(16))) _Float16 v16h;
typedef __attribute__((ext_vector_type(8)))  float    v8f;

#define B_N   32
#define CH    256
#define HW    3136   // 56*56
#define KTOT  2304   // 256*9
#define WPS   589824 // 256*256*9 weights per sample per layer

union Frag16 { uint4 q[2]; v16h h; };   // 32B WMMA A/B operand
union AccU   { v8f v; float f[8]; };    // 16x16 f32 C/D slice per lane
union OutV   { _Float16 h[8]; uint4 q; };

// ---------------------------------------------------------------------------
// 1) x [32][256][3136] f32  ->  x_t [32][3136][256] f16  (LDS tile transpose)
// grid (49, 8, 32), block 256
// ---------------------------------------------------------------------------
__global__ __launch_bounds__(256) void k_transpose_x(
    const float* __restrict__ x, _Float16* __restrict__ xt) {
  __shared__ _Float16 tile[32][68];
  const int b  = blockIdx.z;
  const int c0 = blockIdx.y * 32;
  const int p0 = blockIdx.x * 64;
  const int t  = threadIdx.x;
  for (int r = 0; r < 8; ++r) {              // read coalesced along pixels
    int idx = r * 256 + t;
    int c = idx >> 6, p = idx & 63;
    tile[c][p] = (_Float16)x[((size_t)b * CH + c0 + c) * HW + p0 + p];
  }
  __syncthreads();
  for (int r = 0; r < 8; ++r) {              // write contiguous along channels
    int idx = r * 256 + t;
    int p = idx >> 5, c = idx & 31;
    xt[((size_t)b * HW + p0 + p) * CH + c0 + c] = tile[c][p];
  }
}

// ---------------------------------------------------------------------------
// 1b) zero page for async im2col halo reads (512 B)
// ---------------------------------------------------------------------------
__global__ void k_zero_page(_Float16* __restrict__ z) {
  z[threadIdx.x] = (_Float16)0.f;            // 1 block x 256 threads
}

// ---------------------------------------------------------------------------
// 2) global average pool: gap_raw[b][c] = mean_hw x[b][c][:]
// grid (256, 32), block 256
// ---------------------------------------------------------------------------
__global__ __launch_bounds__(256) void k_gap(
    const float* __restrict__ x, float* __restrict__ gap_raw) {
  const int b = blockIdx.y, c = blockIdx.x;
  const float* p = x + ((size_t)b * CH + c) * HW;
  float s = 0.f;
  for (int i = threadIdx.x; i < HW; i += 256) s += p[i];
  __shared__ float red[256];
  red[threadIdx.x] = s;
  __syncthreads();
  for (int off = 128; off > 0; off >>= 1) {
    if (threadIdx.x < off) red[threadIdx.x] += red[threadIdx.x + off];
    __syncthreads();
  }
  if (threadIdx.x == 0) gap_raw[b * CH + c] = red[0] * (1.0f / (float)HW);
}

// ---------------------------------------------------------------------------
// 3) gap16[b][j] = gap_raw[b] . reduce_w[j] + reduce_b[j]    (32*16 outputs)
// ---------------------------------------------------------------------------
__global__ __launch_bounds__(256) void k_gap_fc(
    const float* __restrict__ gap_raw, const float* __restrict__ rw,
    const float* __restrict__ rb, float* __restrict__ gap16) {
  int i = blockIdx.x * 256 + threadIdx.x;
  if (i >= B_N * 16) return;
  int b = i >> 4, j = i & 15;
  const float* g = gap_raw + b * CH;
  const float* w = rw + j * CH;
  float s = rb[j];
  for (int c = 0; c < CH; ++c) s += g[c] * w[c];
  gap16[i] = s;
}

// ---------------------------------------------------------------------------
// 4) a[b][r] = sigmoid(gap16[b] . fc1_w[r] + fc1_b[r])       (32*4096 outputs)
// grid 512, block 256
// ---------------------------------------------------------------------------
__global__ __launch_bounds__(256) void k_fc1(
    const float* __restrict__ gap16, const float* __restrict__ fw,
    const float* __restrict__ fb, float* __restrict__ a) {
  int i = blockIdx.x * 256 + threadIdx.x;      // 0..131071
  int b = i >> 12, r = i & 4095;
  const float* g = gap16 + b * 16;
  const float* w = fw + r * 16;
  float s = fb[r];
  for (int k = 0; k < 16; ++k) s += g[k] * w[k];
  a[i] = 1.0f / (1.0f + __expf(-s));
}

// ---------------------------------------------------------------------------
// 5) per-sample weight generation -> fp16, layout [b][co][q][ci]
//    flat j = co*2304 + ci*9 + q is also the fc2_w row index; group g = j/576
// grid 73728, block 256
// ---------------------------------------------------------------------------
__global__ __launch_bounds__(256) void k_genw(
    const float* __restrict__ a, const float* __restrict__ fc2,
    _Float16* __restrict__ wt) {
  long long idx = (long long)blockIdx.x * 256 + threadIdx.x;
  if (idx >= (long long)B_N * WPS) return;
  int b = (int)(idx / WPS);
  int j = (int)(idx % WPS);                    // co*2304 + ci*9 + q
  int co = j / KTOT;
  int rem = j % KTOT;
  int ci = rem / 9;
  int q  = rem % 9;
  int g  = j / 576;
  const float4 av = *(const float4*)(a + (size_t)b * 4096 + g * 4);
  const float4 wv = *(const float4*)(fc2 + (size_t)j * 4);
  float v = av.x * wv.x + av.y * wv.y + av.z * wv.z + av.w * wv.w;
  wt[(((size_t)b * CH + co) * 9 + q) * CH + ci] = (_Float16)v;
}

// ---------------------------------------------------------------------------
// CDNA5 async memory->LDS copy (16B per lane, ASYNCcnt-tracked).
// LDS address comes per lane from the vdst VGPR; generic LDS ptr low 32 bits = offset.
// ---------------------------------------------------------------------------
__device__ __forceinline__ void async_copy_b128(void* lds_dst, const void* gsrc) {
  unsigned lds = (unsigned)(uintptr_t)lds_dst;
  unsigned long long ga = (unsigned long long)(uintptr_t)gsrc;
  asm volatile("global_load_async_to_lds_b128 %0, %1, off"
               :: "v"(lds), "v"(ga) : "memory");
}

// ---------------------------------------------------------------------------
// 6) Implicit-GEMM 3x3 conv on WMMA, double-buffered async-LDS pipeline.
//    Per block: sample b, 128co x 128px tile; 8 waves, each owns 32co x 64px
//    = 2x4 WMMA f32 accumulators. 72 K-steps of 32 (9 taps x 8 ci-chunks).
//    Each wave issues exactly 4 async b128 instructions per step (halo rows
//    redirected to a zeroed page so EXEC stays full and ASYNCcnt is exact).
//    LAST=false: BN+ReLU -> fp16 transposed [b][pix][co] (feeds conv2).
//    LAST=true : BN + residual(x) + ReLU -> f32 NCHW d_out.
// grid (25, 2, 32), block 256
// ---------------------------------------------------------------------------
template <bool LAST>
__global__ __launch_bounds__(256) void k_conv_wmma(
    const _Float16* __restrict__ xin,   // [32][3136][256] f16
    const _Float16* __restrict__ wgen,  // [32][256][9][256] f16
    const _Float16* __restrict__ zpage, // 256 zero halves
    const float* __restrict__ bng, const float* __restrict__ bnb,
    const float* __restrict__ bnm, const float* __restrict__ bnv,
    const float* __restrict__ resid,    // x f32 NCHW (LAST only)
    _Float16* __restrict__ out_h,       // !LAST: [b][pix][co] f16
    float* __restrict__ out_f)          // LAST : [b][co][pix] f32
{
  const int b   = blockIdx.z;
  const int co0 = blockIdx.y * 128;
  const int p0  = blockIdx.x * 128;
  const int t    = threadIdx.x;
  const int lane = t & 31;
  const int wave = t >> 5;
  const int wr   = wave >> 1;          // 0..3 : 32-co sub-tile
  const int wc   = wave & 1;           // 0..1 : 64-px sub-tile
  const int half = lane >> 4;          // K-half selector per ISA layout
  const int l16  = lane & 15;

  // double-buffered: A rows 128 x 64B, B cols 128 x 64B per buffer (32 KB)
  __shared__ uint4 As[2][512];
  __shared__ uint4 Bs[2][512];

  v8f acc[2][4] = {};

  const _Float16* wbase = wgen + ((size_t)b * CH + co0) * KTOT;
  const _Float16* xbase = xin + (size_t)b * HW * CH;

  // per-thread fixed segment coordinates (2 x 16B segments of each tile)
  const int segA = t * 2;

  auto issue = [&](int step, int buf) {
    const int q = step >> 3, ck = step & 7;
    const int kh = q / 3 - 1, kw = q % 3 - 1;
#pragma unroll
    for (int r = 0; r < 2; ++r) {               // A: weights, always in-bounds
      int seg = segA + r;
      int row = seg >> 2, s4 = seg & 3;
      const _Float16* src = wbase + (size_t)row * KTOT + q * 256 + ck * 32 + s4 * 8;
      async_copy_b128(&As[buf][row * 4 + s4], src);
    }
#pragma unroll
    for (int r = 0; r < 2; ++r) {               // B: im2col, halo -> zero page
      int seg = segA + r;
      int i = seg >> 2, s4 = seg & 3;
      int p = p0 + i;
      int y = p / 56, xx = p % 56;
      int ys = y + kh, xs = xx + kw;
      const _Float16* src =
          (p < HW && (unsigned)ys < 56u && (unsigned)xs < 56u)
              ? xbase + (size_t)(ys * 56 + xs) * CH + ck * 32 + s4 * 8
              : zpage + s4 * 8;
      async_copy_b128(&Bs[buf][i * 4 + s4], src);
    }
  };

  issue(0, 0);                                   // prologue: fill buffer 0

  for (int s = 0; s < 72; ++s) {
    const int buf = s & 1;
    if (s < 71) {
      issue(s + 1, buf ^ 1);                     // prefetch next K-step
      asm volatile("s_wait_asynccnt 4" ::: "memory");  // batch s landed (own wave)
    } else {
      asm volatile("s_wait_asynccnt 0" ::: "memory");
    }
    __syncthreads();                             // batch s landed (all waves)

    // ---- operand fetch matching CDNA5 WMMA VGPR layouts ----
    Frag16 afr[2], bfr[4];
#pragma unroll
    for (int m = 0; m < 2; ++m) {
      int row = wr * 32 + m * 16 + l16;          // A: lane = M row
      afr[m].q[0] = As[buf][row * 4 + half];     // K 0-7  / 8-15
      afr[m].q[1] = As[buf][row * 4 + half + 2]; // K 16-23 / 24-31
    }
#pragma unroll
    for (int n = 0; n < 4; ++n) {
      int col = wc * 64 + n * 16 + l16;          // B: lane = N col
      bfr[n].q[0] = Bs[buf][col * 4 + half];     // K 0-7  / 8-15
      bfr[n].q[1] = Bs[buf][col * 4 + half + 2]; // K 16-23 / 24-31
    }
#pragma unroll
    for (int m = 0; m < 2; ++m)
#pragma unroll
      for (int n = 0; n < 4; ++n)
        acc[m][n] = __builtin_amdgcn_wmma_f32_16x16x32_f16(
            false, afr[m].h, false, bfr[n].h, (short)0, acc[m][n],
            false, false);
    __syncthreads();                             // reads done before overwrite
  }

  // ---- fused epilogue. C layout: lane l16 = N, VGPR r -> M = r + half*8 ----
  for (int m = 0; m < 2; ++m) {
    const int cobase = co0 + wr * 32 + m * 16 + half * 8;  // 8 consecutive co
    float sc[8], sh[8];
    for (int r = 0; r < 8; ++r) {
      int c = cobase + r;
      float s = bng[c] * rsqrtf(bnv[c] + 1e-5f);
      sc[r] = s;
      sh[r] = bnb[c] - bnm[c] * s;
    }
    for (int n = 0; n < 4; ++n) {
      const int p = p0 + wc * 64 + n * 16 + l16;
      if (p >= HW) continue;
      AccU u; u.v = acc[m][n];
      if (!LAST) {
        OutV ov;
        for (int r = 0; r < 8; ++r) {
          float v = u.f[r] * sc[r] + sh[r];
          ov.h[r] = (_Float16)(v > 0.f ? v : 0.f);
        }
        *(uint4*)(out_h + ((size_t)b * HW + p) * CH + cobase) = ov.q;  // 1x b128
      } else {
        for (int r = 0; r < 8; ++r) {
          int c = cobase + r;
          float v = u.f[r] * sc[r] + sh[r];
          v += resid[((size_t)b * CH + c) * HW + p];
          out_f[((size_t)b * CH + c) * HW + p] = v > 0.f ? v : 0.f;
        }
      }
    }
  }
}

// ---------------------------------------------------------------------------
extern "C" void kernel_launch(void* const* d_in, const int* in_sizes, int n_in,
                              void* d_out, int out_size, void* d_ws,
                              size_t ws_size, hipStream_t stream) {
  const float* x        = (const float*)d_in[0];
  const float* reduce_w = (const float*)d_in[1];
  const float* reduce_b = (const float*)d_in[2];
  const float* w1_fc1_w = (const float*)d_in[3];
  const float* w1_fc1_b = (const float*)d_in[4];
  const float* w1_fc2_w = (const float*)d_in[5];
  const float* bn1_g = (const float*)d_in[6];
  const float* bn1_b = (const float*)d_in[7];
  const float* bn1_m = (const float*)d_in[8];
  const float* bn1_v = (const float*)d_in[9];
  const float* w2_fc1_w = (const float*)d_in[10];
  const float* w2_fc1_b = (const float*)d_in[11];
  const float* w2_fc2_w = (const float*)d_in[12];
  const float* bn2_g = (const float*)d_in[13];
  const float* bn2_b = (const float*)d_in[14];
  const float* bn2_m = (const float*)d_in[15];
  const float* bn2_v = (const float*)d_in[16];
  float* out = (float*)d_out;

  // workspace carve-up (~141 MB; activations+weights per layer fit in 192MB L2)
  char* ws = (char*)d_ws;
  const size_t XT_BYTES = (size_t)B_N * HW * CH * sizeof(_Float16);  // 51.4 MB
  const size_t WT_BYTES = (size_t)B_N * WPS * sizeof(_Float16);      // 37.7 MB
  _Float16* x_t  = (_Float16*)ws;
  _Float16* y1_t = (_Float16*)(ws + XT_BYTES);
  _Float16* w_t  = (_Float16*)(ws + 2 * XT_BYTES);                   // reused L1/L2
  float* gap_raw = (float*)(ws + 2 * XT_BYTES + WT_BYTES);
  float* gap16   = gap_raw + B_N * CH;
  float* a_buf   = gap16 + B_N * 16;
  _Float16* zpage = (_Float16*)(a_buf + B_N * 4096);                 // 512 B zeros

  // prep: zero page, fp16 transpose of x, GAP bottleneck vector
  k_zero_page<<<1, 256, 0, stream>>>(zpage);
  k_transpose_x<<<dim3(49, 8, 32), 256, 0, stream>>>(x, x_t);
  k_gap<<<dim3(256, 32), 256, 0, stream>>>(x, gap_raw);
  k_gap_fc<<<2, 256, 0, stream>>>(gap_raw, reduce_w, reduce_b, gap16);

  const dim3 cgrid(25, 2, 32);   // 128px x 128co tiles, 32 samples

  // layer 1: gen weights -> WMMA conv -> BN+ReLU (fp16, transposed)
  k_fc1<<<512, 256, 0, stream>>>(gap16, w1_fc1_w, w1_fc1_b, a_buf);
  k_genw<<<(B_N * WPS) / 256, 256, 0, stream>>>(a_buf, w1_fc2_w, w_t);
  k_conv_wmma<false><<<cgrid, 256, 0, stream>>>(
      x_t, w_t, zpage, bn1_g, bn1_b, bn1_m, bn1_v, nullptr, y1_t, nullptr);

  // layer 2: gen weights -> WMMA conv -> BN + residual + ReLU (f32 NCHW)
  k_fc1<<<512, 256, 0, stream>>>(gap16, w2_fc1_w, w2_fc1_b, a_buf);
  k_genw<<<(B_N * WPS) / 256, 256, 0, stream>>>(a_buf, w2_fc2_w, w_t);
  k_conv_wmma<true><<<cgrid, 256, 0, stream>>>(
      y1_t, w_t, zpage, bn2_g, bn2_b, bn2_m, bn2_v, x, nullptr, out);
}
